// PerformerBlock_58652073394310
// MI455X (gfx1250) — matrix operand order repaired and checked by reading, not compiled, not run on hardware
//
#include <hip/hip_runtime.h>
#include <hip/hip_bf16.h>
#include <stdint.h>

// ---------------------------------------------------------------------------
// Performer (linear attention) transformer block for MI455X (gfx1250, wave32)
// GEMMs via v_wmma_f32_16x16x32_bf16 with pre-packed bf16 B-fragments.
// FF kernel stages its A-tile to LDS with the Tensor Data Mover (TDM).
// ---------------------------------------------------------------------------

typedef __bf16 bf16_t;
typedef __attribute__((ext_vector_type(16))) __bf16 v16bf;
typedef __attribute__((ext_vector_type(8)))  float  v8f;
typedef __attribute__((ext_vector_type(4)))  unsigned int u32x4;
typedef __attribute__((ext_vector_type(8)))  int i32x8;
typedef __attribute__((ext_vector_type(4)))  int i32x4;

#if __has_builtin(__builtin_amdgcn_tensor_load_to_lds)
#define HAVE_TDM 1
#endif

#define D_MODEL 256
#define NHEAD   8
#define D_HEAD  32
#define DIM_FF  1024
#define BATCH   4
#define SEQ     16384
#define NTOK    (BATCH * SEQ)
#define KV_SPLIT 64

// ---- WMMA fragment helpers (wave32; A and B both use contiguous k = 16*(lane/16) + s) ----------

__device__ __forceinline__ int a_koff(int r, int hi) {
  return (hi << 4) + (r << 1);
}

__device__ __forceinline__ v16bf frag_a_f32(const float* __restrict__ A, int lda,
                                            int row0, int kk) {
  int lane = threadIdx.x & 31;
  int m = lane & 15, hi = lane >> 4;
  const float* p = A + (size_t)(row0 + m) * lda + kk;
  v16bf a;
#pragma unroll
  for (int r = 0; r < 8; ++r) {
    int k0 = a_koff(r, hi);
    a[2 * r]     = (bf16_t)p[k0];
    a[2 * r + 1] = (bf16_t)p[k0 + 1];
  }
  return a;
}

__device__ __forceinline__ v16bf frag_a_bf16(const bf16_t* __restrict__ A, int lda,
                                             int row0, int kk) {
  int lane = threadIdx.x & 31;
  int m = lane & 15, hi = lane >> 4;
  const bf16_t* p = A + (size_t)(row0 + m) * lda + kk;
  v16bf a;
#pragma unroll
  for (int r = 0; r < 8; ++r) {
    int k0 = a_koff(r, hi);
    a[2 * r]     = p[k0];
    a[2 * r + 1] = p[k0 + 1];
  }
  return a;
}

// A-fragment gathered from an LDS-resident [16][ld] bf16 tile (ds_load path)
__device__ __forceinline__ v16bf frag_a_lds(const bf16_t* base, int ld, int kk) {
  int lane = threadIdx.x & 31;
  int m = lane & 15, hi = lane >> 4;
  const bf16_t* p = base + m * ld + kk;
  v16bf a;
#pragma unroll
  for (int r = 0; r < 8; ++r) {
    int k0 = a_koff(r, hi);
    a[2 * r]     = p[k0];
    a[2 * r + 1] = p[k0 + 1];
  }
  return a;
}

__device__ __forceinline__ v16bf frag_b_f32(const float* __restrict__ Bm, int ldb,
                                            int kk, int col0) {
  int lane = threadIdx.x & 31;
  int n = lane & 15;
  int kb = (lane >> 4) << 4;
  const float* p = Bm + (size_t)(kk + kb) * ldb + col0 + n;
  v16bf b;
#pragma unroll
  for (int r = 0; r < 8; ++r) {
    b[2 * r]     = (bf16_t)p[(size_t)(2 * r) * ldb];
    b[2 * r + 1] = (bf16_t)p[(size_t)(2 * r + 1) * ldb];
  }
  return b;
}

// Packed-B loader: [tile][lane][16 bf16] -> one 32B contiguous load per lane.
__device__ __forceinline__ v16bf frag_b_pk(const bf16_t* __restrict__ P,
                                           int ntiles, int kk, int col0) {
  int lane = threadIdx.x & 31;
  size_t tile = (size_t)(kk >> 5) * ntiles + (col0 >> 4);
  return *(const v16bf*)(P + tile * 512 + lane * 16);
}

__device__ __forceinline__ v8f wmma_bf16(v16bf a, v16bf b, v8f c) {
  return __builtin_amdgcn_wmma_f32_16x16x32_bf16(false, a, false, b, (short)0, c,
                                                 false, false);
}

// ---------------------------------------------------------------------------
// Weight packing: f32 [K x N] row-major -> bf16 B-fragment tiles.
// ---------------------------------------------------------------------------
__global__ __launch_bounds__(32) void pack_b_kernel(const float* __restrict__ W,
                                                    bf16_t* __restrict__ out,
                                                    int N) {
  int nt = N >> 4;
  int kk   = (blockIdx.x / nt) << 5;
  int col0 = (blockIdx.x % nt) << 4;
  v16bf b = frag_b_f32(W, N, kk, col0);
  int lane = threadIdx.x & 31;
  *(v16bf*)(out + (size_t)blockIdx.x * 512 + lane * 16) = b;
}

// ---------------------------------------------------------------------------
// Kernel 1: QKV projection + phi. 8 waves; hoisted A, 6 col tiles per wave.
// ---------------------------------------------------------------------------
__global__ __launch_bounds__(256) void qkv_phi_kernel(
    const float* __restrict__ x,
    const bf16_t* __restrict__ pWq, const float* __restrict__ bq,
    const bf16_t* __restrict__ pWk, const float* __restrict__ bk,
    const bf16_t* __restrict__ pWv, const float* __restrict__ bv,
    bf16_t* __restrict__ qph, bf16_t* __restrict__ kph, bf16_t* __restrict__ vout)
{
  int row0 = blockIdx.x << 4;
  int wave = threadIdx.x >> 5;
  int lane = threadIdx.x & 31;
  int n = lane & 15, hi = lane >> 4;

  v16bf afr[8];
#pragma unroll
  for (int q = 0; q < 8; ++q) afr[q] = frag_a_f32(x, D_MODEL, row0, q * 32);

  for (int i = 0; i < 6; ++i) {
    int j = wave + (i << 3);        // 0..47
    int mat  = j >> 4;              // 0=Q,1=K,2=V
    int col0 = (j & 15) << 4;
    const bf16_t* pW  = (mat == 0) ? pWq : (mat == 1) ? pWk : pWv;
    const float* bias = (mat == 0) ? bq  : (mat == 1) ? bk  : bv;
    bf16_t* out       = (mat == 0) ? qph : (mat == 1) ? kph : vout;

    v8f acc = {};
#pragma unroll
    for (int q = 0; q < 8; ++q) {
      v16bf b = frag_b_pk(pW, 16, q * 32, col0);
      acc = wmma_bf16(afr[q], b, acc);
    }
    float bn = bias[col0 + n];
#pragma unroll
    for (int r = 0; r < 8; ++r) {
      int m = r + (hi << 3);
      float v = acc[r] + bn;
      if (mat < 2) v = (v > 0.f) ? (v + 1.f) : __expf(v);   // phi = elu+1
      out[(size_t)(row0 + m) * D_MODEL + col0 + n] = (bf16_t)v;
    }
  }
}

// ---------------------------------------------------------------------------
__global__ void zero_kernel(float* __restrict__ p, int n) {
  int i = blockIdx.x * blockDim.x + threadIdx.x;
  if (i < n) p[i] = 0.f;
}

// ---------------------------------------------------------------------------
// Kernel 3: kv = sum_t phi(k)^T v ; k_sum via ones-vector WMMA.
// ---------------------------------------------------------------------------
__global__ __launch_bounds__(32) void kv_reduce_kernel(
    const bf16_t* __restrict__ kph, const bf16_t* __restrict__ vph,
    float* __restrict__ kv, float* __restrict__ ksum)
{
  int bh = blockIdx.x;
  int part = blockIdx.y;
  int b = bh >> 3, h = bh & 7;
  const int tokens_per = SEQ / KV_SPLIT;
  size_t t0 = (size_t)b * SEQ + (size_t)part * tokens_per;
  int lane = threadIdx.x & 31;
  int hcol = h * D_HEAD;
  int m = lane & 15, hi = lane >> 4;
  int n = lane & 15, kb = (lane >> 4) << 4;

  v16bf ones;
#pragma unroll
  for (int i = 0; i < 16; ++i) ones[i] = (bf16_t)1.0f;

  v8f a00 = {}, a01 = {}, a10 = {}, a11 = {};
  v8f s0 = {}, s1 = {};

  for (int c = 0; c < tokens_per; c += 32) {
    size_t tc = t0 + c;
    v16bf fa0, fa1;
#pragma unroll
    for (int r = 0; r < 8; ++r) {
      int kt = a_koff(r, hi);
      const bf16_t* p0 = kph + (tc + kt) * D_MODEL + hcol + m;
      const bf16_t* p1 = kph + (tc + kt + 1) * D_MODEL + hcol + m;
      fa0[2 * r] = p0[0];  fa0[2 * r + 1] = p1[0];
      fa1[2 * r] = p0[16]; fa1[2 * r + 1] = p1[16];
    }
    v16bf fb0, fb1;
#pragma unroll
    for (int r = 0; r < 8; ++r) {
      const bf16_t* q0 = vph + (tc + kb + 2 * r) * D_MODEL + hcol + n;
      const bf16_t* q1 = vph + (tc + kb + 2 * r + 1) * D_MODEL + hcol + n;
      fb0[2 * r] = q0[0];  fb0[2 * r + 1] = q1[0];
      fb1[2 * r] = q0[16]; fb1[2 * r + 1] = q1[16];
    }
    a00 = wmma_bf16(fa0, fb0, a00);
    a01 = wmma_bf16(fa0, fb1, a01);
    a10 = wmma_bf16(fa1, fb0, a10);
    a11 = wmma_bf16(fa1, fb1, a11);
    s0  = wmma_bf16(fa0, ones, s0);
    s1  = wmma_bf16(fa1, ones, s1);
  }

  float* kvb = kv + (size_t)bh * (32 * 32);
#pragma unroll
  for (int r = 0; r < 8; ++r) {
    int mm = r + (hi << 3);
    atomicAdd(kvb + (size_t)mm * 32 + n,             a00[r]);
    atomicAdd(kvb + (size_t)mm * 32 + 16 + n,        a01[r]);
    atomicAdd(kvb + (size_t)(mm + 16) * 32 + n,      a10[r]);
    atomicAdd(kvb + (size_t)(mm + 16) * 32 + 16 + n, a11[r]);
  }
  if (n == 0) {
#pragma unroll
    for (int r = 0; r < 8; ++r) {
      int mm = r + (hi << 3);
      atomicAdd(ksum + (size_t)bh * 32 + mm,      s0[r]);
      atomicAdd(ksum + (size_t)bh * 32 + 16 + mm, s1[r]);
    }
  }
}

// ---------------------------------------------------------------------------
// Kernel 4: attn = z * (phi(q) @ kv_h)
// ---------------------------------------------------------------------------
__global__ __launch_bounds__(256) void attn_out_kernel(
    const bf16_t* __restrict__ qph, const float* __restrict__ kv,
    const float* __restrict__ ksum, bf16_t* __restrict__ attn)
{
  __shared__ float zsh[16][NHEAD];
  int row0 = blockIdx.x << 4;
  int tid  = threadIdx.x;
  int bb   = row0 / SEQ;

  if (tid < 16 * NHEAD) {
    int rm = tid & 15, h = tid >> 4;
    const bf16_t* qp = qph + (size_t)(row0 + rm) * D_MODEL + h * D_HEAD;
    const float* kp  = ksum + (size_t)(bb * NHEAD + h) * 32;
    float s = 0.f;
#pragma unroll
    for (int d = 0; d < 32; ++d) s += (float)qp[d] * kp[d];
    zsh[rm][h] = 1.f / (s + 1e-6f);
  }
  __syncthreads();

  int wave = tid >> 5;
  int lane = tid & 31;
  int m = lane & 15, hi = lane >> 4;
  int n = lane & 15, kb = (lane >> 4) << 4;

#pragma unroll
  for (int i = 0; i < 2; ++i) {
    int col0 = ((wave << 1) + i) << 4;
    int h  = col0 >> 5;
    int e0 = col0 & 31;
    int bh = bb * NHEAD + h;
    int hcol = h * D_HEAD;

    v16bf a;
    {
      const bf16_t* p = qph + (size_t)(row0 + m) * D_MODEL + hcol;
#pragma unroll
      for (int r = 0; r < 8; ++r) {
        int k0 = a_koff(r, hi);
        a[2 * r] = p[k0];  a[2 * r + 1] = p[k0 + 1];
      }
    }
    const float* kvb = kv + (size_t)bh * (32 * 32);
    v16bf bf;
#pragma unroll
    for (int r = 0; r < 8; ++r) {
      bf[2 * r]     = (bf16_t)kvb[(size_t)(kb + 2 * r) * 32 + e0 + n];
      bf[2 * r + 1] = (bf16_t)kvb[(size_t)(kb + 2 * r + 1) * 32 + e0 + n];
    }
    v8f acc = {};
    acc = wmma_bf16(a, bf, acc);

#pragma unroll
    for (int r = 0; r < 8; ++r) {
      int mm = r + (hi << 3);
      attn[(size_t)(row0 + mm) * D_MODEL + col0 + n] =
          (bf16_t)(acc[r] * zsh[mm][h]);
    }
  }
}

// ---------------------------------------------------------------------------
// Kernel 5: s = attn @ Wo + bo + x (residual)
// ---------------------------------------------------------------------------
__global__ __launch_bounds__(256) void out_proj_kernel(
    const bf16_t* __restrict__ attn, const bf16_t* __restrict__ pWo,
    const float* __restrict__ bo, const float* __restrict__ x,
    float* __restrict__ s)
{
  int row0 = blockIdx.x << 4;
  int wave = threadIdx.x >> 5;
  int lane = threadIdx.x & 31;
  int n = lane & 15, hi = lane >> 4;

  v16bf afr[8];
#pragma unroll
  for (int q = 0; q < 8; ++q) afr[q] = frag_a_bf16(attn, D_MODEL, row0, q * 32);

#pragma unroll
  for (int i = 0; i < 2; ++i) {
    int col0 = (wave + (i << 3)) << 4;
    v8f acc = {};
#pragma unroll
    for (int q = 0; q < 8; ++q) {
      v16bf b = frag_b_pk(pWo, 16, q * 32, col0);
      acc = wmma_bf16(afr[q], b, acc);
    }
    float bn = bo[col0 + n];
#pragma unroll
    for (int r = 0; r < 8; ++r) {
      int m = r + (hi << 3);
      size_t idx = (size_t)(row0 + m) * D_MODEL + col0 + n;
      s[idx] = acc[r] + bn + x[idx];
    }
  }
}

// ---------------------------------------------------------------------------
// Kernel 6: LayerNorm. wave32 shuffle reduction + 8-entry LDS combine.
// ---------------------------------------------------------------------------
__global__ __launch_bounds__(256) void layernorm_kernel(
    const float* __restrict__ s, const float* __restrict__ g,
    const float* __restrict__ bta, float* __restrict__ outf,
    bf16_t* __restrict__ outb)
{
  int t = blockIdx.x;
  int i = threadIdx.x;
  __shared__ float part[8];
  float v = s[(size_t)t * D_MODEL + i];

  float sum = v;
#pragma unroll
  for (int mk = 16; mk >= 1; mk >>= 1) sum += __shfl_xor(sum, mk, 32);
  if ((i & 31) == 0) part[i >> 5] = sum;
  __syncthreads();
  float tot = 0.f;
#pragma unroll
  for (int w = 0; w < 8; ++w) tot += part[w];
  float mu = tot * (1.f / D_MODEL);

  float d = v - mu;
  float sq = d * d;
#pragma unroll
  for (int mk = 16; mk >= 1; mk >>= 1) sq += __shfl_xor(sq, mk, 32);
  __syncthreads();
  if ((i & 31) == 0) part[i >> 5] = sq;
  __syncthreads();
  float tv = 0.f;
#pragma unroll
  for (int w = 0; w < 8; ++w) tv += part[w];
  float var = tv * (1.f / D_MODEL);

  float y = d * rsqrtf(var + 1e-5f) * g[i] + bta[i];
  outf[(size_t)t * D_MODEL + i] = y;
  if (outb) outb[(size_t)t * D_MODEL + i] = (bf16_t)y;
}

// ---------------------------------------------------------------------------
// Kernel 7: fused FF: s = x1 + relu(x1@W1+b1)@W2 + b2
// x1 A-tile staged to LDS via TDM (tensor_load_to_lds + s_wait_tensorcnt);
// hidden 16x1024 bf16 in padded LDS; packed weights.
// ---------------------------------------------------------------------------
#define FF_PAD 8
__global__ __launch_bounds__(256) void ff_kernel(
    const bf16_t* __restrict__ x1b, const float* __restrict__ x1f,
    const bf16_t* __restrict__ pW1, const float* __restrict__ b1,
    const bf16_t* __restrict__ pW2, const float* __restrict__ b2,
    float* __restrict__ s)
{
  __shared__ bf16_t atile[16 * D_MODEL];          // 8 KB: x1 A-tile
  __shared__ bf16_t hsh[16][DIM_FF + FF_PAD];     // 33 KB: hidden tile
  int row0 = blockIdx.x << 4;
  int wave = threadIdx.x >> 5;
  int lane = threadIdx.x & 31;
  int n = lane & 15, hi = lane >> 4;

#ifdef HAVE_TDM
  // Wave 0 issues one TDM descriptor: 2-D tile, 16 rows x 256 bf16, into LDS.
  // D# fields per CDNA5 ISA.
  // amdgpu-toolchain 6-arg form: (u32x4, i32x8, i32x4, i32x4, i32x8, i32 cpol)
  if (wave == 0) {
    unsigned lds_base = (unsigned)(uintptr_t)(&atile[0]);   // LDS offset = addr[31:0]
    unsigned long long ga =
        (unsigned long long)(uintptr_t)(x1b + (size_t)row0 * D_MODEL);
    u32x4 g0;
    g0[0] = 1u;                                             // count=1, user D#
    g0[1] = lds_base;                                       // lds_addr
    g0[2] = (unsigned)ga;                                   // global_addr[31:0]
    g0[3] = ((unsigned)(ga >> 32) & 0x01FFFFFFu) | 0x80000000u; // addr[56:32], type=2
    i32x8 g1;
    g1[0] = 0x00010000;            // workgroup_mask=0, data_size=1 (2 bytes)
    g1[1] = (int)(256u << 16);     // tensor_dim0 = 256 (low 16 in [31:16])
    g1[2] = 0;                     // dim0 hi=0, tensor_dim1 low16 = 0
    g1[3] = (int)(1u | (256u << 16)); // tensor_dim1 hi16=1 (65536), tile_dim0=256
    g1[4] = 16;                    // tile_dim1 = 16, tile_dim2 = 0
    g1[5] = 256;                   // tensor_dim0_stride low32 = 256
    g1[6] = (int)(256u << 16);     // stride0 hi=0, tensor_dim1_stride low16 = 256
    g1[7] = 0;                     // stride1 hi = 0
    i32x4 g2 = {0, 0, 0, 0}, g3 = {0, 0, 0, 0};
    i32x8 g4 = {0, 0, 0, 0, 0, 0, 0, 0};
    __builtin_amdgcn_tensor_load_to_lds(g0, g1, g2, g3, g4, 0);
    __builtin_amdgcn_s_wait_tensorcnt((short)0);            // TENSORcnt == 0
  }
  __syncthreads();
#else
  for (int i = threadIdx.x; i < 16 * D_MODEL; i += 256)
    atile[i] = x1b[(size_t)row0 * D_MODEL + i];
  __syncthreads();
#endif

  // A fragments from the LDS-resident tile (ds_load path)
  v16bf afr[8];
#pragma unroll
  for (int q = 0; q < 8; ++q) afr[q] = frag_a_lds(atile, D_MODEL, q * 32);

  // Phase 1: hidden = relu(x1 @ W1 + b1) -> LDS (bf16)
  for (int jt = wave; jt < DIM_FF / 16; jt += 8) {
    int col0 = jt << 4;
    v8f acc = {};
#pragma unroll
    for (int q = 0; q < 8; ++q) {
      v16bf b = frag_b_pk(pW1, DIM_FF / 16, q * 32, col0);
      acc = wmma_bf16(afr[q], b, acc);
    }
    float bn = b1[col0 + n];
#pragma unroll
    for (int r = 0; r < 8; ++r) {
      int m = r + (hi << 3);
      float v = acc[r] + bn;
      hsh[m][col0 + n] = (bf16_t)(v > 0.f ? v : 0.f);
    }
  }
  __syncthreads();

  // Phase 2: out = hidden @ W2 + b2 + x1
  for (int jt = wave; jt < D_MODEL / 16; jt += 8) {
    int col0 = jt << 4;
    v8f acc = {};
    for (int kk = 0; kk < DIM_FF; kk += 32) {
      int m = lane & 15;
      v16bf a;
#pragma unroll
      for (int r = 0; r < 8; ++r) {
        int k0 = kk + a_koff(r, hi);
        a[2 * r] = hsh[m][k0];  a[2 * r + 1] = hsh[m][k0 + 1];
      }
      v16bf b = frag_b_pk(pW2, D_MODEL / 16, kk, col0);
      acc = wmma_bf16(a, b, acc);
    }
    float bn = b2[col0 + n];
#pragma unroll
    for (int r = 0; r < 8; ++r) {
      int m = r + (hi << 3);
      size_t idx = (size_t)(row0 + m) * D_MODEL + col0 + n;
      s[idx] = acc[r] + bn + x1f[idx];
    }
  }
}

// ---------------------------------------------------------------------------
// Host launcher
// ---------------------------------------------------------------------------
extern "C" void kernel_launch(void* const* d_in, const int* in_sizes, int n_in,
                              void* d_out, int out_size, void* d_ws, size_t ws_size,
                              hipStream_t stream)
{
  const float* x   = (const float*)d_in[0];
  const float* Wq  = (const float*)d_in[1];
  const float* bq  = (const float*)d_in[2];
  const float* Wk  = (const float*)d_in[3];
  const float* bk  = (const float*)d_in[4];
  const float* Wv  = (const float*)d_in[5];
  const float* bv  = (const float*)d_in[6];
  const float* Wo  = (const float*)d_in[7];
  const float* bo  = (const float*)d_in[8];
  const float* W1  = (const float*)d_in[9];
  const float* b1  = (const float*)d_in[10];
  const float* W2  = (const float*)d_in[11];
  const float* b2  = (const float*)d_in[12];
  const float* g1  = (const float*)d_in[13];
  const float* be1 = (const float*)d_in[14];
  const float* g2  = (const float*)d_in[15];
  const float* be2 = (const float*)d_in[16];

  char* ws = (char*)d_ws;
  const size_t SZ_BF = (size_t)NTOK * D_MODEL * sizeof(bf16_t);   // 33.5 MB
  const size_t SZ_F  = (size_t)NTOK * D_MODEL * sizeof(float);    // 67 MB

  bf16_t* qph  = (bf16_t*)(ws);
  bf16_t* kph  = (bf16_t*)(ws + SZ_BF);               // k_phi; reused as attn
  bf16_t* vph  = (bf16_t*)(ws + 2 * SZ_BF);           // v; reused as x1 bf16
  float*  kv   = (float*)(ws + 3 * SZ_BF);
  float*  ksum = kv + BATCH * NHEAD * 32 * 32;
  char* p2 = ws + 3 * SZ_BF +
             (BATCH * NHEAD * 32 * 32 + BATCH * NHEAD * 32) * sizeof(float);
  p2 = (char*)(((size_t)p2 + 255) & ~(size_t)255);
  float*  sbuf = (float*)p2;
  float*  x1f  = (float*)(p2 + SZ_F);
  char* p3 = p2 + 2 * SZ_F;
  bf16_t* pWq = (bf16_t*)p3;
  bf16_t* pWk = pWq + D_MODEL * D_MODEL;
  bf16_t* pWv = pWk + D_MODEL * D_MODEL;
  bf16_t* pWo = pWv + D_MODEL * D_MODEL;
  bf16_t* pW1 = pWo + D_MODEL * D_MODEL;
  bf16_t* pW2 = pW1 + D_MODEL * DIM_FF;
  bf16_t* attn = kph;
  bf16_t* x1b  = vph;

  const int ROWT = NTOK / 16;

  pack_b_kernel<<<(D_MODEL / 32) * (D_MODEL / 16), 32, 0, stream>>>(Wq, pWq, D_MODEL);
  pack_b_kernel<<<(D_MODEL / 32) * (D_MODEL / 16), 32, 0, stream>>>(Wk, pWk, D_MODEL);
  pack_b_kernel<<<(D_MODEL / 32) * (D_MODEL / 16), 32, 0, stream>>>(Wv, pWv, D_MODEL);
  pack_b_kernel<<<(D_MODEL / 32) * (D_MODEL / 16), 32, 0, stream>>>(Wo, pWo, D_MODEL);
  pack_b_kernel<<<(D_MODEL / 32) * (DIM_FF / 16), 32, 0, stream>>>(W1, pW1, DIM_FF);
  pack_b_kernel<<<(DIM_FF / 32) * (D_MODEL / 16), 32, 0, stream>>>(W2, pW2, D_MODEL);

  qkv_phi_kernel<<<ROWT, 256, 0, stream>>>(x, pWq, bq, pWk, bk, pWv, bv,
                                           qph, kph, vph);
  {
    int nz = BATCH * NHEAD * 32 * 32 + BATCH * NHEAD * 32;
    zero_kernel<<<(nz + 255) / 256, 256, 0, stream>>>(kv, nz);
  }
  kv_reduce_kernel<<<dim3(BATCH * NHEAD, KV_SPLIT), 32, 0, stream>>>(kph, vph,
                                                                     kv, ksum);
  attn_out_kernel<<<ROWT, 256, 0, stream>>>(qph, kv, ksum, attn);
  out_proj_kernel<<<ROWT, 256, 0, stream>>>(attn, pWo, bo, x, sbuf);
  layernorm_kernel<<<NTOK, 256, 0, stream>>>(sbuf, g1, be1, x1f, x1b);
  ff_kernel<<<ROWT, 256, 0, stream>>>(x1b, x1f, pW1, b1, pW2, b2, sbuf);
  layernorm_kernel<<<NTOK, 256, 0, stream>>>(sbuf, g2, be2, (float*)d_out,
                                             (bf16_t*)nullptr);
}
